// ListenAndDecideWithContext_63488206569864
// MI455X (gfx1250) — matrix operand order repaired and checked by reading, not compiled, not run on hardware
//
#include <hip/hip_runtime.h>
#include <hip/hip_bf16.h>
#include <stdint.h>

// ---------------------------------------------------------------------------
// ListenAndDecideWithContext — MI455X (gfx1250, wave32, WMMA)
//
// Shapes: B=256, C=256, E=128, F=512, H=1024, N=B*C=65536 rows.
// Pipeline:
//   prep:      W1 -> W1T (H x F, bf16), W2 -> W2T (E x H, bf16),
//              fold b1 + BN(eval) into per-H scale/bias.
//   norm_msg:  row-normalize embedded msg -> a_norm (N x 128, bf16)
//   mlp:       per 32-row block: X->LDS(bf16) -> WMMA GEMM1 (K=512) ->
//              scale/bias+ReLU -> H in LDS(bf16) -> WMMA GEMM2 (K=1024) ->
//              logits in LDS(f32) -> row-normalize -> b_norm (bf16, global)
//   sim:       per batch: async-DMA stage a_norm/b_norm tiles into LDS
//              (global_load_async_to_lds_b128 + s_wait_asynccnt), then
//              WMMA a @ b^T -> (256x256) * (1/T)
// All matmuls use v_wmma_f32_16x16x32_bf16 (f32 accumulate).
// ---------------------------------------------------------------------------

typedef __attribute__((ext_vector_type(16))) __bf16 v16bf;
typedef __attribute__((ext_vector_type(8)))  __bf16 v8bf;
typedef __attribute__((ext_vector_type(4)))  __bf16 v4bf;
typedef __attribute__((ext_vector_type(8)))  float  v8f;

#define NB   256           // batches
#define NC   256           // contexts / messages per batch
#define NE   128           // embedding dim
#define NF   512           // context feature dim
#define NH   1024          // hidden dim
#define NROWS (NB * NC)    // 65536
#define BN_EPS 1e-5f
#define COS_EPS 1e-8f
#define INV_T (1.0f / 0.07f)

// ---- WMMA operand loaders (row-major bf16 sources) -------------------------
// A-matrix 16x32 (MxK), ISA layout: lane<16 -> row=lane, K {0..7, 16..23};
// lane>=16 -> row=lane-16, K {8..15, 24..31}.
__device__ __forceinline__ v16bf load_A16(const __bf16* base, int ld, int lane) {
    const int m  = lane & 15;
    const int k0 = (lane & 16) ? 8 : 0;
    const __bf16* p = base + m * ld + k0;
    v8bf lo = *(const v8bf*)(p);          // K k0 .. k0+7
    v8bf hi = *(const v8bf*)(p + 16);     // K k0+16 .. k0+23
    return __builtin_shufflevector(lo, hi, 0,1,2,3,4,5,6,7,8,9,10,11,12,13,14,15);
}

// B-matrix 32x16 (KxN), same K order as A: lane<16 -> col=lane, K {0..7, 16..23};
// lane>=16 -> col=lane-16, K {8..15, 24..31}. base points at row-major
// B^T (N x K), so per lane this is two 16-byte chunks.
__device__ __forceinline__ v16bf load_B16(const __bf16* base, int ld, int lane) {
    const int n  = lane & 15;
    const int k0 = (lane & 16) ? 8 : 0;
    const __bf16* p = base + n * ld + k0;
    v8bf lo = *(const v8bf*)(p);
    v8bf hi = *(const v8bf*)(p + 16);
    return __builtin_shufflevector(lo, hi, 0,1,2,3,4,5,6,7,8,9,10,11,12,13,14,15);
}

__device__ __forceinline__ v8f wmma_bf16(v16bf a, v16bf b, v8f c) {
    return __builtin_amdgcn_wmma_f32_16x16x32_bf16(false, a, false, b,
                                                   (short)0, c, false, false);
}

// Async DMA: global -> LDS, 16 bytes per lane (GV mode), ASYNCcnt-tracked.
__device__ __forceinline__ void async_g2l_b128(uint32_t lds_off, const void* gptr) {
    uint64_t ga = (uint64_t)(uintptr_t)gptr;
    asm volatile("global_load_async_to_lds_b128 %0, %1, off"
                 :: "v"(lds_off), "v"(ga) : "memory");
}
__device__ __forceinline__ void wait_asynccnt0() {
    asm volatile("s_wait_asynccnt 0x0" ::: "memory");
}

// ---- prep: transpose weights to bf16, fold BN ------------------------------
__global__ void prep_kernel(const float* __restrict__ W1, const float* __restrict__ W2,
                            const float* __restrict__ b1, const float* __restrict__ gamma,
                            const float* __restrict__ beta, const float* __restrict__ rmean,
                            const float* __restrict__ rvar,
                            __bf16* __restrict__ W1T, __bf16* __restrict__ W2T,
                            float* __restrict__ scaleH, float* __restrict__ biasH) {
    const int tid = blockIdx.x * blockDim.x + threadIdx.x;
    const int stride = gridDim.x * blockDim.x;
    // W1T[n][k] = W1[k][n] : (H x F)
    for (int i = tid; i < NH * NF; i += stride) {
        int n = i / NF, k = i - n * NF;
        W1T[i] = (__bf16)W1[k * NH + n];
    }
    // W2T[e][h] = W2[h][e] : (E x H)
    for (int i = tid; i < NE * NH; i += stride) {
        int e = i / NH, h = i - e * NH;
        W2T[i] = (__bf16)W2[h * NE + e];
    }
    // BN fold: y = acc*s + ((b1-mean)*s + beta), s = gamma*rsqrt(var+eps)
    if (tid < NH) {
        float s = gamma[tid] * rsqrtf(rvar[tid] + BN_EPS);
        scaleH[tid] = s;
        biasH[tid]  = (b1[tid] - rmean[tid]) * s + beta[tid];
    }
}

// ---- normalize embedded message rows (E=128), write bf16 -------------------
__global__ void norm_msg_kernel(const float* __restrict__ msg,
                                __bf16* __restrict__ anorm) {
    const int row  = blockIdx.x * 8 + (threadIdx.x >> 5);   // one wave32 per row
    const int lane = threadIdx.x & 31;
    const float4 v = ((const float4*)(msg + (size_t)row * NE))[lane];
    float ss = v.x * v.x + v.y * v.y + v.z * v.z + v.w * v.w;
    ss += __shfl_xor(ss, 16, 32);
    ss += __shfl_xor(ss,  8, 32);
    ss += __shfl_xor(ss,  4, 32);
    ss += __shfl_xor(ss,  2, 32);
    ss += __shfl_xor(ss,  1, 32);
    const float inv = 1.0f / fmaxf(sqrtf(ss), COS_EPS);
    v4bf o = { (__bf16)(v.x * inv), (__bf16)(v.y * inv),
               (__bf16)(v.z * inv), (__bf16)(v.w * inv) };
    *(v4bf*)(anorm + (size_t)row * NE + lane * 4) = o;
}

// ---- fused MLP: GEMM1 + BN + ReLU + GEMM2 + row-normalize ------------------
// 32-row block per workgroup, 256 threads (8 wave32s).
#define BM   32
#define LDX  (NF + 8)     // 520  (pad: rows shift 4 banks -> conflict-free)
#define LDH  (NH + 8)     // 1032
#define LDL  (NE + 4)     // 132
__global__ __launch_bounds__(256)
void mlp_kernel(const float* __restrict__ X,
                const __bf16* __restrict__ W1T, const __bf16* __restrict__ W2T,
                const float* __restrict__ scaleH, const float* __restrict__ biasH,
                __bf16* __restrict__ bnorm) {
    __shared__ __align__(16) __bf16 Xs[BM * LDX];   // 33,280 B
    __shared__ __align__(16) __bf16 Hs[BM * LDH];   // 66,048 B
    __shared__ __align__(16) float  Ls[BM * LDL];   // 16,896 B (~113.5 KB tot)

    const int tid  = threadIdx.x;
    const int wave = tid >> 5;
    const int lane = tid & 31;
    const int row0 = blockIdx.x * BM;

    // Phase 0: stage X block (32 x 512 f32) into LDS as bf16, coalesced float4
    const float* Xg = X + (size_t)row0 * NF;
    for (int i = tid; i < (BM * NF) / 4; i += 256) {
        const float4 v = ((const float4*)Xg)[i];
        const int r = i >> 7;              // 128 float4 per row
        const int c = (i & 127) << 2;
        v4bf o = { (__bf16)v.x, (__bf16)v.y, (__bf16)v.z, (__bf16)v.w };
        *(v4bf*)&Xs[r * LDX + c] = o;
    }
    __syncthreads();

    // Phase 1: H = relu(scale * (X @ W1) + bias) ; M=32, N=1024, K=512
    // 2 M-tiles x 64 N-tiles; each wave owns 8 N-tiles.
    for (int j = 0; j < 8; ++j) {
        const int nt = wave * 8 + j;
        const float s  = scaleH[nt * 16 + (lane & 15)];
        const float bb = biasH [nt * 16 + (lane & 15)];
        for (int mt = 0; mt < 2; ++mt) {
            v8f acc = {0.f, 0.f, 0.f, 0.f, 0.f, 0.f, 0.f, 0.f};
#pragma unroll 4
            for (int kt = 0; kt < NF / 32; ++kt) {
                v16bf a = load_A16(&Xs[(mt * 16) * LDX + kt * 32], LDX, lane);
                v16bf b = load_B16(W1T + (size_t)(nt * 16) * NF + kt * 32, NF, lane);
                acc = wmma_bf16(a, b, acc);
            }
            const int col = nt * 16 + (lane & 15);
            const int moff = mt * 16 + ((lane >> 4) << 3);
#pragma unroll
            for (int r = 0; r < 8; ++r) {
                float y = fmaxf(acc[r] * s + bb, 0.0f);
                Hs[(moff + r) * LDH + col] = (__bf16)y;
            }
        }
    }
    __syncthreads();

    // Phase 2: logits = H @ W2 ; M=32, N=128, K=1024 -> 16 tiles, 2 per wave
    for (int t = 0; t < 2; ++t) {
        const int id = wave * 2 + t;
        const int mt = id >> 3, nt = id & 7;
        v8f acc = {0.f, 0.f, 0.f, 0.f, 0.f, 0.f, 0.f, 0.f};
#pragma unroll 4
        for (int kt = 0; kt < NH / 32; ++kt) {
            v16bf a = load_A16(&Hs[(mt * 16) * LDH + kt * 32], LDH, lane);
            v16bf b = load_B16(W2T + (size_t)(nt * 16) * NH + kt * 32, NH, lane);
            acc = wmma_bf16(a, b, acc);
        }
        const int col  = nt * 16 + (lane & 15);
        const int moff = mt * 16 + ((lane >> 4) << 3);
#pragma unroll
        for (int r = 0; r < 8; ++r)
            Ls[(moff + r) * LDL + col] = acc[r];
    }
    __syncthreads();

    // Phase 3: row-normalize logits, emit bf16 b_norm. 4 rows per wave.
    for (int i = 0; i < 4; ++i) {
        const int r = wave * 4 + i;
        const float4 v = *(const float4*)&Ls[r * LDL + lane * 4];
        float ss = v.x * v.x + v.y * v.y + v.z * v.z + v.w * v.w;
        ss += __shfl_xor(ss, 16, 32);
        ss += __shfl_xor(ss,  8, 32);
        ss += __shfl_xor(ss,  4, 32);
        ss += __shfl_xor(ss,  2, 32);
        ss += __shfl_xor(ss,  1, 32);
        const float inv = 1.0f / fmaxf(sqrtf(ss), COS_EPS);
        v4bf o = { (__bf16)(v.x * inv), (__bf16)(v.y * inv),
                   (__bf16)(v.z * inv), (__bf16)(v.w * inv) };
        *(v4bf*)(bnorm + (size_t)(row0 + r) * NE + lane * 4) = o;
    }
}

// ---- batched similarity: sim[b] = (1/T) * a_norm[b] @ b_norm[b]^T ----------
// Workgroup: 128 threads (4 waves), 64x64 output tile. Both 64x128 bf16 tiles
// are staged into LDS with the async DMA engine (ASYNCcnt), then consumed
// with bank-conflict-free ds_load_b128 (row stride 136 bf16 = 68 dwords).
#define LDT (NE + 8)   // 136
__global__ __launch_bounds__(128)
void sim_kernel(const __bf16* __restrict__ anorm, const __bf16* __restrict__ bnorm,
                float* __restrict__ out) {
    __shared__ __align__(16) __bf16 As[64 * LDT];   // 17,408 B
    __shared__ __align__(16) __bf16 Bs[64 * LDT];   // 17,408 B

    const int batch = blockIdx.y;
    const int tm = (blockIdx.x >> 2) * 64;
    const int tn = (blockIdx.x &  3) * 64;
    const int t    = threadIdx.x;
    const int wave = t >> 5;
    const int lane = t & 31;

    const __bf16* Ag = anorm + (size_t)batch * NC * NE + (size_t)tm * NE;
    const __bf16* Bg = bnorm + (size_t)batch * NC * NE + (size_t)tn * NE;

    // Stage: 64 rows x 128 bf16 = 1024 chunks of 16B per tile; 8 per thread.
#pragma unroll
    for (int i = 0; i < 8; ++i) {
        const int id  = t + 128 * i;     // 0..1023
        const int row = id >> 4;
        const int c8  = (id & 15) << 3;  // bf16 element offset (16B chunks)
        async_g2l_b128((uint32_t)(uintptr_t)&As[row * LDT + c8], Ag + row * NE + c8);
        async_g2l_b128((uint32_t)(uintptr_t)&Bs[row * LDT + c8], Bg + row * NE + c8);
    }
    wait_asynccnt0();
    __syncthreads();

    const int mrow = wave * 16;
    v8f acc0 = {0.f,0.f,0.f,0.f,0.f,0.f,0.f,0.f};
    v8f acc1 = acc0, acc2 = acc0, acc3 = acc0;

#pragma unroll
    for (int kt = 0; kt < NE / 32; ++kt) {
        v16bf a  = load_A16(&As[mrow * LDT + kt * 32], LDT, lane);
        v16bf b0 = load_B16(&Bs[ 0 * LDT + kt * 32], LDT, lane);
        v16bf b1 = load_B16(&Bs[16 * LDT + kt * 32], LDT, lane);
        v16bf b2 = load_B16(&Bs[32 * LDT + kt * 32], LDT, lane);
        v16bf b3 = load_B16(&Bs[48 * LDT + kt * 32], LDT, lane);
        acc0 = wmma_bf16(a, b0, acc0);
        acc1 = wmma_bf16(a, b1, acc1);
        acc2 = wmma_bf16(a, b2, acc2);
        acc3 = wmma_bf16(a, b3, acc3);
    }

    // C/D layout: VGPR r, lane l -> row mrow + r + 8*(l>=16), col l%16
    float* O = out + (size_t)batch * NC * NC;
    const int rbase = tm + mrow + ((lane >> 4) << 3);
    const int cbase = tn + (lane & 15);
#pragma unroll
    for (int r = 0; r < 8; ++r) {
        O[(size_t)(rbase + r) * NC + cbase +  0] = acc0[r] * INV_T;
        O[(size_t)(rbase + r) * NC + cbase + 16] = acc1[r] * INV_T;
        O[(size_t)(rbase + r) * NC + cbase + 32] = acc2[r] * INV_T;
        O[(size_t)(rbase + r) * NC + cbase + 48] = acc3[r] * INV_T;
    }
}

// ---- host launch -----------------------------------------------------------
extern "C" void kernel_launch(void* const* d_in, const int* in_sizes, int n_in,
                              void* d_out, int out_size, void* d_ws, size_t ws_size,
                              hipStream_t stream) {
    (void)in_sizes; (void)n_in; (void)out_size; (void)ws_size;
    const float* msg   = (const float*)d_in[0];   // (B,C,E)
    const float* X     = (const float*)d_in[1];   // (B,C,F)
    const float* W1    = (const float*)d_in[2];   // (F,H)
    const float* b1    = (const float*)d_in[3];   // (H)
    const float* gam   = (const float*)d_in[4];   // (H)
    const float* bet   = (const float*)d_in[5];   // (H)
    const float* rmean = (const float*)d_in[6];   // (H)
    const float* rvar  = (const float*)d_in[7];   // (H)
    const float* W2    = (const float*)d_in[8];   // (H,E)
    float* out = (float*)d_out;                   // (B,C,C)

    char* ws = (char*)d_ws;
    constexpr size_t OFF_W1T   = 0;                          // 1,048,576 B
    constexpr size_t OFF_W2T   = 1u << 20;                   //   262,144 B
    constexpr size_t OFF_SCALE = OFF_W2T + (256u << 10);     //     4,096 B
    constexpr size_t OFF_BIAS  = OFF_SCALE + 4096;           //     4,096 B
    constexpr size_t OFF_ANORM = 2u << 20;                   // 16,777,216 B
    constexpr size_t OFF_BNORM = OFF_ANORM + (16u << 20);    // 16,777,216 B

    __bf16* W1T    = (__bf16*)(ws + OFF_W1T);
    __bf16* W2T    = (__bf16*)(ws + OFF_W2T);
    float*  scaleH = (float*) (ws + OFF_SCALE);
    float*  biasH  = (float*) (ws + OFF_BIAS);
    __bf16* anorm  = (__bf16*)(ws + OFF_ANORM);
    __bf16* bnorm  = (__bf16*)(ws + OFF_BNORM);

    prep_kernel<<<512, 256, 0, stream>>>(W1, W2, b1, gam, bet, rmean, rvar,
                                         W1T, W2T, scaleH, biasH);
    norm_msg_kernel<<<NROWS / 8, 256, 0, stream>>>(msg, anorm);
    mlp_kernel<<<NROWS / BM, 256, 0, stream>>>(X, W1T, W2T, scaleH, biasH, bnorm);
    sim_kernel<<<dim3(16, NB), 128, 0, stream>>>(anorm, bnorm, out);
}
